// InjectedEncoder_5068061409850
// MI455X (gfx1250) — matrix operand order repaired and checked by reading, not compiled, not run on hardware
//
#include <hip/hip_runtime.h>
#include <cstring>

typedef __attribute__((ext_vector_type(16))) _Float16 v16h;
typedef __attribute__((ext_vector_type(4)))  _Float16 v4h;
typedef __attribute__((ext_vector_type(8)))  float    v8f;

namespace {
constexpr int kB     = 8;
constexpr int kHW    = 64 * 64;
constexpr int kNF    = 32;
constexpr int kG     = 4;
constexpr int kF     = 384;
constexpr int kCG    = 96;
constexpr int kLAT   = 512;
constexpr int kDKH   = 1024;
constexpr int kS     = kG * kCG * kCG;    // 36864
constexpr int kDKT   = 3 * kS + 3 * kF;   // 111744
constexpr int kOUTC  = 97;
constexpr int kSTATE = 8224;
constexpr int kNC    = 8;
}

__device__ __forceinline__ float leaky(float v) { return v > 0.f ? v : 0.2f * v; }

// ---------------------------------------------------------------------------
// Generic batched GEMM:  C[M,N] = epilogue( A[M,K] @ B[K,N] )
//  * fp16 WMMA (v_wmma_f32_16x16x32_f16), f32 accumulate
//  * REQUIRES N % 128 == 0 and K % 32 == 0 (true for every GEMM in this net)
//    -> no K/N bounds checks, unconditional b128 global loads, b64 LDS stores
//  * M may be ragged (M = 8, 97); handled branchlessly: A row index is clamped
//    to a valid row, loaded unconditionally, then zeroed by select (no
//    exec-mask predication, no forced s_wait_loadcnt serialization)
//  * INACT : leaky-ReLU applied to B elements on load (activation-as-input)
//  * BROW  : bias indexed by output row (conv channel) instead of column
//  * GATE  : C = Res * sigmoid(Res + acc + bias)      (dyna gating, in-place)
//  * OUTACT: leaky-ReLU on (acc + bias + beta*Cin) before alpha/residual
//  standard epilogue: C = scaleR*Res + alpha*(acc + bias + beta*Cin)
// Batching: grid.z decoded as b = z / G, g = z % G with per-b / per-g element
// strides for every operand (supports per-sample grouped 1x1 convs).
// ---------------------------------------------------------------------------
struct GemmArgs {
  const float* A;  const float* Bw; const float* bias;
  const float* Cin; const float* Res; float* C;
  long sAb, sAg, sBb, sBg, sbb, sbg, sCib, sCig, sRb, sRg, sCb, sCg;
  int M, N, K, G;
  float alpha, beta, scaleR;
};

template<bool INACT, bool BROW, bool GATE, bool OUTACT>
__global__ __launch_bounds__(256) void k_gemm(GemmArgs p) {
  const int t  = threadIdx.x;
  const int b  = blockIdx.z / p.G, g = blockIdx.z % p.G;
  const int Mb = blockIdx.y * 32, Nb = blockIdx.x * 128;

  const float* A  = p.A  + (long)b * p.sAb + (long)g * p.sAg;
  const float* Bm = p.Bw + (long)b * p.sBb + (long)g * p.sBg;

  __shared__ __align__(16) _Float16 Al[32][40];    // [m][k]
  __shared__ __align__(16) _Float16 Bl[128][40];   // [n][k]  (transposed stage)

  v8f acc0 = {}; v8f acc1 = {};
  const int lane = t & 31, wave = t >> 5;
  const int m0 = (wave >> 2) * 16, n0 = (wave & 3) * 32;   // 2 x 4 wave grid
  const int lg = lane >> 4, lr = lane & 15;

  // staging coordinates
  const int am = t >> 3, ak = (t & 7) * 4;          // A: one float4 per thread
  const int bk = (t >> 5) * 4, bn = (t & 31) * 4;   // B: 4(k) x 4(n) block per thread
  const bool aFull  = (Mb + 32 <= p.M);
  const bool aValid = aFull || (Mb + am) < p.M;     // thread-level validity
  const int  amRow  = aValid ? (Mb + am) : (p.M - 1);  // clamped, always legal

  for (int k0 = 0; k0 < p.K; k0 += 32) {
    // ---- stage A tile (32x32): branchless b128 load + b64 LDS store ----
    float4 a4 = *(const float4*)&A[(long)amRow * p.K + (k0 + ak)];
    if (!aValid) { a4.x = 0.f; a4.y = 0.f; a4.z = 0.f; a4.w = 0.f; }
    v4h ah = {(_Float16)a4.x, (_Float16)a4.y, (_Float16)a4.z, (_Float16)a4.w};
    *(v4h*)&Al[am][ak] = ah;

    // ---- stage B tile (32x128): four b128 loads, register 4x4 transpose,
    //      four b64 LDS stores into [n][k] layout ----
    const float* bp0 = &Bm[(long)(k0 + bk) * p.N + Nb + bn];
    float4 r0 = *(const float4*)(bp0);
    float4 r1 = *(const float4*)(bp0 + p.N);
    float4 r2 = *(const float4*)(bp0 + 2L * p.N);
    float4 r3 = *(const float4*)(bp0 + 3L * p.N);
    if (INACT) {
      r0.x = leaky(r0.x); r0.y = leaky(r0.y); r0.z = leaky(r0.z); r0.w = leaky(r0.w);
      r1.x = leaky(r1.x); r1.y = leaky(r1.y); r1.z = leaky(r1.z); r1.w = leaky(r1.w);
      r2.x = leaky(r2.x); r2.y = leaky(r2.y); r2.z = leaky(r2.z); r2.w = leaky(r2.w);
      r3.x = leaky(r3.x); r3.y = leaky(r3.y); r3.z = leaky(r3.z); r3.w = leaky(r3.w);
    }
    v4h c0 = {(_Float16)r0.x, (_Float16)r1.x, (_Float16)r2.x, (_Float16)r3.x};
    v4h c1 = {(_Float16)r0.y, (_Float16)r1.y, (_Float16)r2.y, (_Float16)r3.y};
    v4h c2 = {(_Float16)r0.z, (_Float16)r1.z, (_Float16)r2.z, (_Float16)r3.z};
    v4h c3 = {(_Float16)r0.w, (_Float16)r1.w, (_Float16)r2.w, (_Float16)r3.w};
    *(v4h*)&Bl[bn + 0][bk] = c0;
    *(v4h*)&Bl[bn + 1][bk] = c1;
    *(v4h*)&Bl[bn + 2][bk] = c2;
    *(v4h*)&Bl[bn + 3][bk] = c3;

    // prefetch next K-tile of B into the cache hierarchy
    if (k0 + 32 < p.K)
      __builtin_prefetch(&Bm[(long)(k0 + 32 + bk) * p.N + Nb + bn], 0, 0);

    __syncthreads();

    // A fragment: lane (lg,lr): m=lr, VGPR q holds K pair (ISA 7.12.2, 16-bit A 16x32)
    union V16 { unsigned int u[8]; v16h v; };
    V16 av;
    #pragma unroll
    for (int q = 0; q < 8; ++q) {
      int kb2 = lg * 16 + 2 * q;
      av.u[q] = *(const unsigned int*)&Al[m0 + lr][kb2];
    }
    v16h af = av.v;

    #pragma unroll
    for (int bj = 0; bj < 2; ++bj) {
      // B fragment: lane (lg,lr): n=lr, halves h -> k = lg*16 + h (contiguous)
      V16 bv;
      const unsigned int* bq = (const unsigned int*)&Bl[n0 + bj * 16 + lr][lg * 16];
      #pragma unroll
      for (int q = 0; q < 8; ++q) bv.u[q] = bq[q];
      if (bj == 0)
        acc0 = __builtin_amdgcn_wmma_f32_16x16x32_f16(false, af, false, bv.v,
                                                      (short)0, acc0, false, false);
      else
        acc1 = __builtin_amdgcn_wmma_f32_16x16x32_f16(false, af, false, bv.v,
                                                      (short)0, acc1, false, false);
    }
    __syncthreads();
  }

  const float* bias = p.bias ? p.bias + (long)b * p.sbb  + (long)g * p.sbg  : nullptr;
  const float* Cin  = p.Cin  ? p.Cin  + (long)b * p.sCib + (long)g * p.sCig : nullptr;
  const float* Res  = p.Res  ? p.Res  + (long)b * p.sRb  + (long)g * p.sRg  : nullptr;
  float*       C    = p.C    + (long)b * p.sCb  + (long)g * p.sCg;

  #pragma unroll
  for (int bj = 0; bj < 2; ++bj) {
    v8f acc = bj ? acc1 : acc0;
    int gn = Nb + n0 + bj * 16 + lr;            // N tile always full
    #pragma unroll
    for (int r = 0; r < 8; ++r) {
      int gm = Mb + m0 + lg * 8 + r;            // C/D layout: m = r + 8*(lane/16)
      if (!aFull && gm >= p.M) continue;
      float v = acc[r];
      if (bias) v += BROW ? bias[gm] : bias[gn];
      long ci = (long)gm * p.N + gn;
      if (GATE) {
        float rz = Res[ci];
        float d  = rz + v;
        C[ci] = rz * (1.f / (1.f + __expf(-d)));
      } else {
        if (Cin) v += p.beta * Cin[ci];
        if (OUTACT) v = leaky(v);
        float o = p.alpha * v;
        if (Res) o += p.scaleR * Res[ci];
        C[ci] = o;
      }
    }
  }
}

// ---------------------------------------------------------------------------
// helpers
// ---------------------------------------------------------------------------
__device__ inline void blockReduce2(float& a, float& b,
                                    volatile float* rA, volatile float* rB) {
  #pragma unroll
  for (int o = 16; o > 0; o >>= 1) {
    a += __shfl_down(a, o, 32);
    b += __shfl_down(b, o, 32);
  }
  int w = threadIdx.x >> 5, l = threadIdx.x & 31;
  if (l == 0) { rA[w] = a; rB[w] = b; }
  __syncthreads();
  if (threadIdx.x == 0) {
    float sa = 0.f, sb = 0.f;
    for (int i = 0; i < 8; ++i) { sa += rA[i]; sb += rB[i]; }
    rA[0] = sa; rB[0] = sb;
  }
  __syncthreads();
  a = rA[0]; b = rB[0];
  __syncthreads();
}

// fused multi-scale sin-sobel filter bank + per-channel instance norm.
// block = one (b,c) image; 80x80 haloed tile in LDS feeds all 4 scales.
__global__ __launch_bounds__(256) void k_sobel_norm(const float* __restrict__ in,
                                                    float* __restrict__ z) {
  __shared__ float tile[80][81];
  __shared__ float wcs[34], wsn[34];
  __shared__ float rA[8], rB[8];
  const int b = blockIdx.x >> 5;
  const int c = blockIdx.x & 31;
  const int t = threadIdx.x;
  const int cOff[5] = {0, 3, 8, 17, 34};
  const int cKS[4]  = {3, 5, 9, 17};

  if (t < 34) {                                    // 1-D separable weights
    int s = 0;
    while (s < 3 && t >= cOff[s + 1]) ++s;
    int i = t - cOff[s];
    float ci = -1.f + 2.f * i / (float)(cKS[s] - 1);
    wcs[t] = cosf(1.57079632679f * ci);
    wsn[t] = sinf(1.57079632679f * ci);
  }
  const float* src = in + ((long)b * kNF + c) * kHW;
  for (int e = t; e < 6400; e += 256) {            // zero-padded halo load
    int y = e / 80, x = e % 80;
    int iy = y - 8, ix = x - 8;
    float v = 0.f;
    if (iy >= 0 && iy < 64 && ix >= 0 && ix < 64) v = src[iy * 64 + ix];
    tile[y][x] = v;
  }
  __syncthreads();

  // identity channels: one stats pass, 4 normalized copies
  float s1 = 0.f, s2 = 0.f;
  #pragma unroll
  for (int j = 0; j < 16; ++j) {
    int idx = t + j * 256;
    float v = tile[(idx >> 6) + 8][(idx & 63) + 8];
    s1 += v; s2 += v * v;
  }
  blockReduce2(s1, s2, rA, rB);
  {
    float m   = s1 * (1.f / 4096.f);
    float var = fmaxf(s2 * (1.f / 4096.f) - m * m, 0.f);
    float inv = rsqrtf(var + 1e-5f);
    for (int j = 0; j < 16; ++j) {
      int idx = t + j * 256;
      float v = (tile[(idx >> 6) + 8][(idx & 63) + 8] - m) * inv;
      #pragma unroll
      for (int s = 0; s < 4; ++s)
        z[((long)b * kF + s * 96 + c) * kHW + idx] = v;
    }
  }

  // gx / gy per scale, normalized in-block before store
  for (int s = 0; s < 4; ++s) {
    const int k = cKS[s], pd = (k - 1) >> 1, wb = cOff[s];
    for (int d = 0; d < 2; ++d) {                 // d=0: gx, d=1: gy (= gx^T)
      float vals[16];
      float a1 = 0.f, a2 = 0.f;
      for (int j = 0; j < 16; ++j) {
        int idx = t + j * 256;
        int h = idx >> 6, w = idx & 63;
        float acc = 0.f;
        for (int i = 0; i < k; ++i) {
          float wi = d ? wsn[wb + i] : wcs[wb + i];
          const float* row = &tile[h + 8 + i - pd][w + 8 - pd];
          for (int jj = 0; jj < k; ++jj) {
            float wj = d ? wcs[wb + jj] : wsn[wb + jj];
            acc += wi * wj * row[jj];
          }
        }
        vals[j] = acc; a1 += acc; a2 += acc * acc;
      }
      blockReduce2(a1, a2, rA, rB);
      float m   = a1 * (1.f / 4096.f);
      float var = fmaxf(a2 * (1.f / 4096.f) - m * m, 0.f);
      float inv = rsqrtf(var + 1e-5f);
      const long zb = ((long)b * kF + s * 96 + 32 + d * 32 + c) * kHW;
      for (int j = 0; j < 16; ++j) {
        int idx = t + j * 256;
        z[zb + idx] = (vals[j] - m) * inv;
      }
    }
  }
}

__global__ __launch_bounds__(256) void k_inconv(const float* __restrict__ x,
                                                const float* __restrict__ w,
                                                const float* __restrict__ bias,
                                                float* __restrict__ emb0) {
  int idx = blockIdx.x * 256 + threadIdx.x;
  if (idx >= kB * kHW) return;
  int b = idx / kHW, hw = idx % kHW;
  float x0 = x[((long)b * 3 + 0) * kHW + hw];
  float x1 = x[((long)b * 3 + 1) * kHW + hw];
  float x2 = x[((long)b * 3 + 2) * kHW + hw];
  for (int o = 0; o < kNF; ++o)
    emb0[((long)b * kNF + o) * kHW + hw] =
        w[o * 3] * x0 + w[o * 3 + 1] * x1 + w[o * 3 + 2] * x2 + bias[o];
}

// multi_cut state pooling: state = [f.mean | fh.mean(w) | fw.mean(h) | hw]
__global__ __launch_bounds__(256) void k_pool(const float* __restrict__ y,
                                              float* __restrict__ st) {
  __shared__ float rA[8], rB[8];
  const int b = blockIdx.x >> 5, c = blockIdx.x & 31;
  const int t = threadIdx.x;
  const float* yb = y + (long)b * kOUTC * kHW;
  float s = 0.f, dummy = 0.f;
  for (int e = t; e < kHW; e += 256) s += yb[(long)c * kHW + e];
  blockReduce2(s, dummy, rA, rB);
  if (t == 0) st[(long)b * kSTATE + c] = s * (1.f / 4096.f);
  if (t < 64) {
    float r = 0.f;
    const float* p = yb + (long)(32 + c) * kHW + t * 64;
    for (int w = 0; w < 64; ++w) r += p[w];
    st[(long)b * kSTATE + 32 + c * 64 + t] = r * (1.f / 64.f);
  }
  if (t < 64) {
    float r = 0.f;
    const float* p = yb + (long)(64 + c) * kHW + t;
    for (int h = 0; h < 64; ++h) r += p[h * 64];
    st[(long)b * kSTATE + 2080 + c * 64 + t] = r * (1.f / 64.f);
  }
  if (c == 0)
    for (int e = t; e < kHW; e += 256)
      st[(long)b * kSTATE + 4128 + e] = yb[(long)96 * kHW + e];
}

__global__ void k_copy(float* dst, const float* src, int n) {
  int i = blockIdx.x * 256 + threadIdx.x;
  if (i < n) dst[i] = src[i];
}

// ---------------------------------------------------------------------------
// host side
// ---------------------------------------------------------------------------
static GemmArgs ga_init() {
  GemmArgs p;
  std::memset(&p, 0, sizeof(p));
  p.alpha = 1.f; p.beta = 0.f; p.scaleR = 1.f; p.G = 1;
  return p;
}

static void launch_gemm(const GemmArgs& p, int nb, bool inact, bool brow,
                        bool gate, bool outact, hipStream_t st) {
  dim3 gr((p.N + 127) / 128, (p.M + 31) / 32, nb), bl(256, 1, 1);
  if (gate)        k_gemm<true,  true,  true,  false><<<gr, bl, 0, st>>>(p);
  else if (inact)  k_gemm<true,  true,  false, false><<<gr, bl, 0, st>>>(p);
  else if (brow)   k_gemm<false, true,  false, false><<<gr, bl, 0, st>>>(p);
  else if (outact) k_gemm<false, false, false, true ><<<gr, bl, 0, st>>>(p);
  else             k_gemm<false, false, false, false><<<gr, bl, 0, st>>>(p);
}

extern "C" void kernel_launch(void* const* d_in, const int* in_sizes, int n_in,
                              void* d_out, int out_size, void* d_ws, size_t ws_size,
                              hipStream_t stream) {
  (void)in_sizes; (void)n_in; (void)out_size; (void)ws_size;
  const float* x          = (const float*)d_in[0];
  const float* inj_lat    = (const float*)d_in[1];
  const float* in_conv_w  = (const float*)d_in[2];
  const float* in_conv_b  = (const float*)d_in[3];
  const float* fl_Ws      = (const float*)d_in[4];
  const float* fl_bs      = (const float*)d_in[5];
  const float* fl_W0      = (const float*)d_in[6];
  const float* fl_b0      = (const float*)d_in[7];
  const float* fl_W1      = (const float*)d_in[8];
  const float* fl_b1      = (const float*)d_in[9];
  const float* dk_Ws      = (const float*)d_in[10];
  const float* dk_bs      = (const float*)d_in[11];
  const float* dk_W0      = (const float*)d_in[12];
  const float* dk_b0      = (const float*)d_in[13];
  const float* dk_W1      = (const float*)d_in[14];
  const float* dk_b1      = (const float*)d_in[15];
  const float* fc_c0w     = (const float*)d_in[16];
  const float* fc_c0b     = (const float*)d_in[17];
  const float* fc_c1w     = (const float*)d_in[18];
  const float* fc_c1b     = (const float*)d_in[19];
  const float* fc_csw     = (const float*)d_in[20];
  const float* fc_csb     = (const float*)d_in[21];
  const float* out_conv_w = (const float*)d_in[22];
  const float* out_conv_b = (const float*)d_in[23];
  const float* otl_w      = (const float*)d_in[24];
  const float* otl_b      = (const float*)d_in[25];

  float* lat_out = (float*)d_out;                       // [8,512]
  float* embs    = (float*)d_out + (size_t)kB * kLAT;   // [9,8,32,64,64]
  const long EMB = (long)kB * kNF * kHW;

  float* w = (float*)d_ws;
  size_t off = 0;
  auto alloc = [&](size_t n) { float* p = w + off; off += (n + 63) & ~size_t(63); return p; };
  float* z     = alloc((size_t)kB * kF * kHW);
  float* h1    = alloc((size_t)kB * kF * kHW);
  float* h2    = alloc((size_t)kB * kF * kHW);
  float* ks    = alloc((size_t)kB * kDKT);
  float* kst   = alloc((size_t)kB * kDKT);
  float* dkh   = alloc((size_t)kB * kDKH);
  float* lat0  = alloc((size_t)kB * kLAT);
  float* lat1  = alloc((size_t)kB * kLAT);
  float* hid   = alloc((size_t)kB * kLAT);
  float* tmp1  = alloc((size_t)kB * kLAT);
  float* ybuf  = alloc((size_t)kB * kOUTC * kHW);
  float* state = alloc((size_t)kB * kSTATE);

  k_copy<<<(kB * kLAT + 255) / 256, 256, 0, stream>>>(lat0, inj_lat, kB * kLAT);
  k_inconv<<<(kB * kHW + 255) / 256, 256, 0, stream>>>(x, in_conv_w, in_conv_b, embs);

  for (int c = 0; c < kNC; ++c) {
    const float* outP = embs + (long)c * EMB;
    float* outN = embs + (long)(c + 1) * EMB;
    float* latI = (c & 1) ? lat1 : lat0;
    float* latO = (c & 1) ? lat0 : lat1;

    // latent LinearResidualBlock: lat += 0.1*(lat@Ws + bs + 0.1*(lrelu(lat@W0+b0)@W1+b1))
    { GemmArgs p = ga_init();
      p.A = latI; p.Bw = fl_W0 + (long)c * kLAT * kLAT; p.bias = fl_b0 + (long)c * kLAT;
      p.C = hid; p.M = kB; p.N = kLAT; p.K = kLAT;
      launch_gemm(p, 1, false, false, false, true, stream); }
    { GemmArgs p = ga_init();
      p.A = hid; p.Bw = fl_W1 + (long)c * kLAT * kLAT; p.bias = fl_b1 + (long)c * kLAT;
      p.C = tmp1; p.M = kB; p.N = kLAT; p.K = kLAT;
      launch_gemm(p, 1, false, false, false, false, stream); }
    { GemmArgs p = ga_init();
      p.A = latI; p.Bw = fl_Ws + (long)c * kLAT * kLAT; p.bias = fl_bs + (long)c * kLAT;
      p.Cin = tmp1; p.beta = 0.1f; p.Res = latI; p.scaleR = 1.f; p.alpha = 0.1f;
      p.C = latO; p.M = kB; p.N = kLAT; p.K = kLAT;
      launch_gemm(p, 1, false, false, false, false, stream); }

    // fused sin-sobel + instance norm -> z [8,384,64,64]
    k_sobel_norm<<<kB * kNF, 256, 0, stream>>>(outP, z);

    // dynamic kernel generation: ks = lat@dk_Ws + bs + 0.1*(lrelu(lat@W0+b0)@W1+b1)
    { GemmArgs p = ga_init();
      p.A = latO; p.Bw = dk_W0; p.bias = dk_b0;
      p.C = dkh; p.M = kB; p.N = kDKH; p.K = kLAT;
      launch_gemm(p, 1, false, false, false, true, stream); }
    { GemmArgs p = ga_init();
      p.A = dkh; p.Bw = dk_W1; p.bias = dk_b1;
      p.C = kst; p.M = kB; p.N = kDKT; p.K = kDKH;
      launch_gemm(p, 1, false, false, false, false, stream); }
    { GemmArgs p = ga_init();
      p.A = latO; p.Bw = dk_Ws; p.bias = dk_bs;
      p.Cin = kst; p.beta = 0.1f;
      p.C = ks; p.M = kB; p.N = kDKT; p.K = kLAT;
      launch_gemm(p, 1, false, false, false, false, stream); }

    // DynaResidualBlock: 3 grouped per-sample GEMMs (grid.z = b*4+g)
    { GemmArgs p = ga_init(); p.G = kG;
      p.A = ks; p.sAb = kDKT; p.sAg = kCG * kCG;
      p.Bw = z; p.sBb = (long)kF * kHW; p.sBg = (long)kCG * kHW;
      p.bias = ks + 3 * kS; p.sbb = kDKT; p.sbg = kCG;
      p.C = h1; p.sCb = (long)kF * kHW; p.sCg = (long)kCG * kHW;
      p.M = kCG; p.N = kHW; p.K = kCG;
      launch_gemm(p, kB * kG, false, true, false, false, stream); }
    { GemmArgs p = ga_init(); p.G = kG;
      p.A = ks + kS; p.sAb = kDKT; p.sAg = kCG * kCG;
      p.Bw = h1; p.sBb = (long)kF * kHW; p.sBg = (long)kCG * kHW;
      p.bias = ks + 3 * kS + kF; p.sbb = kDKT; p.sbg = kCG;
      p.C = h2; p.sCb = (long)kF * kHW; p.sCg = (long)kCG * kHW;
      p.M = kCG; p.N = kHW; p.K = kCG;
      launch_gemm(p, kB * kG, true, true, false, false, stream); }
    { GemmArgs p = ga_init(); p.G = kG;       // fused z*sigmoid(z + dyna) gate, in-place
      p.A = ks + 2 * kS; p.sAb = kDKT; p.sAg = kCG * kCG;
      p.Bw = h2; p.sBb = (long)kF * kHW; p.sBg = (long)kCG * kHW;
      p.bias = ks + 3 * kS + 2 * kF; p.sbb = kDKT; p.sbg = kCG;
      p.Res = z; p.sRb = (long)kF * kHW; p.sRg = (long)kCG * kHW;
      p.C = z; p.sCb = (long)kF * kHW; p.sCg = (long)kCG * kHW;
      p.M = kCG; p.N = kHW; p.K = kCG;
      launch_gemm(p, kB * kG, true, true, true, false, stream); }

    // ResidualBlock (1x1 convs) + out update, written straight into emb slot c+1
    { GemmArgs p = ga_init();
      p.A = fc_c0w; p.Bw = z; p.sBb = (long)kF * kHW; p.bias = fc_c0b;
      p.C = h1; p.sCb = (long)kF * kHW;
      p.M = kF; p.N = kHW; p.K = kF;
      launch_gemm(p, kB, true, true, false, false, stream); }
    { GemmArgs p = ga_init();
      p.A = fc_c1w; p.Bw = h1; p.sBb = (long)kF * kHW; p.bias = fc_c1b;
      p.C = h2; p.sCb = (long)kNF * kHW;
      p.M = kNF; p.N = kHW; p.K = kF;
      launch_gemm(p, kB, true, true, false, false, stream); }
    { GemmArgs p = ga_init();   // out_next = out + 0.1*(csw@z + csb + 0.1*dx)
      p.A = fc_csw; p.Bw = z; p.sBb = (long)kF * kHW; p.bias = fc_csb;
      p.Cin = h2; p.sCib = (long)kNF * kHW; p.beta = 0.1f;
      p.Res = outP; p.sRb = (long)kNF * kHW; p.scaleR = 1.f; p.alpha = 0.1f;
      p.C = outN; p.sCb = (long)kNF * kHW;
      p.M = kNF; p.N = kHW; p.K = kF;
      launch_gemm(p, kB, false, true, false, false, stream); }
  }

  // head: y = conv1x1(out, out_conv) ; state pooling ; lat_out = state@otl + b
  { GemmArgs p = ga_init();
    p.A = out_conv_w; p.Bw = embs + (long)kNC * EMB; p.sBb = (long)kNF * kHW;
    p.bias = out_conv_b;
    p.C = ybuf; p.sCb = (long)kOUTC * kHW;
    p.M = kOUTC; p.N = kHW; p.K = kNF;
    launch_gemm(p, kB, false, true, false, false, stream); }
  k_pool<<<kB * kNF, 256, 0, stream>>>(ybuf, state);
  { GemmArgs p = ga_init();
    p.A = state; p.Bw = otl_w; p.bias = otl_b;
    p.C = lat_out; p.M = kB; p.N = kLAT; p.K = kSTATE;
    launch_gemm(p, 1, false, false, false, false, stream); }
}
